// OptimizedTokenToStateRouter_584115553023
// MI455X (gfx1250) — matrix operand order repaired and checked by reading, not compiled, not run on hardware
//
#include <hip/hip_runtime.h>
#include <math.h>

// ---------------- problem constants ----------------
#define B_    4
#define S_    4096
#define N_    128
#define TD    1024        // TOKEN_DIM
#define SD    512         // STATE_DIM
#define HEADS 4
#define HD    128         // HEAD_DIM
#define PD    512         // PROJ_DIM
#define LN_EPS 1e-5f
#define ENT_W  0.01f

#define ROWS_T (B_*S_)            // 16384 token rows
#define ROWS_K (B_*N_)            // 512 state rows
#define TOK_CNT ((size_t)ROWS_T*TD)  // 16777216 floats
#define RW_CNT  ((size_t)ROWS_T*N_)  // 2097152 floats

// ---------------- vector types ----------------
typedef __attribute__((ext_vector_type(8)))  unsigned short u16x8;
typedef __attribute__((ext_vector_type(16))) unsigned short u16x16;
typedef __attribute__((ext_vector_type(16))) __bf16         v16bf;
typedef __attribute__((ext_vector_type(8)))  float          v8f;

// ---------------- helpers ----------------
__device__ __forceinline__ unsigned short f2bf(float f) {
    unsigned int u = __builtin_bit_cast(unsigned int, f);
    unsigned int r = u + 0x7FFFu + ((u >> 16) & 1u);   // round-to-nearest-even
    return (unsigned short)(r >> 16);
}

__device__ __forceinline__ v16bf pack_frag(u16x8 lo, u16x8 hi) {
    u16x16 u;
#pragma unroll
    for (int i = 0; i < 8; ++i) { u[i] = lo[i]; u[8 + i] = hi[i]; }
    return __builtin_bit_cast(v16bf, u);
}

__device__ __forceinline__ v8f wmma_bf16(v16bf a, v16bf b, v8f c) {
    // D(16x16,f32) = A(16x32,bf16) * B(32x16,bf16) + C
    return __builtin_amdgcn_wmma_f32_16x16x32_bf16(
        /*neg_a=*/false, a, /*neg_b=*/false, b,
        /*c_mod=*/(short)0, c, /*reuse_a=*/false, /*reuse_b=*/false);
}

// CDNA5 async global->LDS copy (16 bytes per active lane), tracked by ASYNCcnt.
// VDST operand = VGPR holding LDS byte offset (low 32 bits of flat shared addr).
__device__ __forceinline__ void async_ld16(unsigned int lds_off,
                                           const unsigned short* gptr) {
    asm volatile("global_load_async_to_lds_b128 %0, %1, off"
                 :: "v"(lds_off), "v"(gptr)
                 : "memory");
}
__device__ __forceinline__ void wait_async0() {
    asm volatile("s_wait_asynccnt 0" ::: "memory");
}

__device__ __forceinline__ unsigned int lds_off_of(const void* p) {
    return (unsigned int)(unsigned long long)p;   // addr[31:0] == LDS offset
}

// ---------------- kernel 1: Wq -> bf16, transposed to [N=512][K=1024] ----------------
__global__ void wq_transpose_bf16_kernel(const float* __restrict__ Wq,
                                         unsigned short* __restrict__ wqT) {
    int idx = blockIdx.x * blockDim.x + threadIdx.x;       // over 512*1024
    if (idx >= PD * TD) return;
    int n = idx / TD;
    int k = idx % TD;
    wqT[idx] = f2bf(Wq[(size_t)k * PD + n]);               // Wq stored [in,out]
}

// ---------------- kernel 2: LayerNorm tokens + passthrough copy ----------------
// one block (256 threads) per token row of 1024
__global__ void ln_tokens_kernel(const float* __restrict__ tokens,
                                 const float* __restrict__ g,
                                 const float* __restrict__ b,
                                 float* __restrict__ out_tokens,
                                 unsigned short* __restrict__ tn) {
    __shared__ float sA[256];
    __shared__ float sB[256];
    const int row = blockIdx.x;
    const int tid = threadIdx.x;
    const float* x = tokens + (size_t)row * TD;

    float4 v = *(const float4*)(x + tid * 4);
    // passthrough copy of raw tokens to output
    *(float4*)(out_tokens + (size_t)row * TD + tid * 4) = v;

    float s  = v.x + v.y + v.z + v.w;
    float s2 = v.x*v.x + v.y*v.y + v.z*v.z + v.w*v.w;
    sA[tid] = s; sB[tid] = s2;
    __syncthreads();
    for (int off = 128; off > 0; off >>= 1) {
        if (tid < off) { sA[tid] += sA[tid + off]; sB[tid] += sB[tid + off]; }
        __syncthreads();
    }
    const float mu   = sA[0] * (1.0f / TD);
    const float var  = sB[0] * (1.0f / TD) - mu * mu;
    const float rstd = rsqrtf(var + LN_EPS);

    int col = tid * 4;
    float4 gg = *(const float4*)(g + col);
    float4 bb = *(const float4*)(b + col);
    ushort4 o;
    o.x = f2bf((v.x - mu) * rstd * gg.x + bb.x);
    o.y = f2bf((v.y - mu) * rstd * gg.y + bb.y);
    o.z = f2bf((v.z - mu) * rstd * gg.z + bb.z);
    o.w = f2bf((v.w - mu) * rstd * gg.w + bb.w);
    *(ushort4*)(tn + (size_t)row * TD + col) = o;
}

// ---------------- kernel 3: LayerNorm states + K projection (small) ----------------
// one block (128 threads) per state row; k stored bf16 row-major [b*128+n][512]
__global__ void k_prep_kernel(const float* __restrict__ states,
                              const float* __restrict__ g,
                              const float* __restrict__ b,
                              const float* __restrict__ Wk,
                              const float* __restrict__ bk,
                              unsigned short* __restrict__ kmat) {
    __shared__ float sA[128];
    __shared__ float sB[128];
    __shared__ float xn[SD];
    const int row = blockIdx.x;        // 0 .. 511
    const int tid = threadIdx.x;
    const float* x = states + (size_t)row * SD;

    float4 v = *(const float4*)(x + tid * 4);
    float s  = v.x + v.y + v.z + v.w;
    float s2 = v.x*v.x + v.y*v.y + v.z*v.z + v.w*v.w;
    sA[tid] = s; sB[tid] = s2;
    __syncthreads();
    for (int off = 64; off > 0; off >>= 1) {
        if (tid < off) { sA[tid] += sA[tid + off]; sB[tid] += sB[tid + off]; }
        __syncthreads();
    }
    const float mu   = sA[0] * (1.0f / SD);
    const float var  = sB[0] * (1.0f / SD) - mu * mu;
    const float rstd = rsqrtf(var + LN_EPS);

    int col = tid * 4;
    float4 gg = *(const float4*)(g + col);
    float4 bb = *(const float4*)(b + col);
    xn[col + 0] = (v.x - mu) * rstd * gg.x + bb.x;
    xn[col + 1] = (v.y - mu) * rstd * gg.y + bb.y;
    xn[col + 2] = (v.z - mu) * rstd * gg.z + bb.z;
    xn[col + 3] = (v.w - mu) * rstd * gg.w + bb.w;
    __syncthreads();

    // each thread computes 4 consecutive outputs j = tid*4 .. tid*4+3
    const int j = tid * 4;
    float4 acc = *(const float4*)(bk + j);
#pragma unroll 4
    for (int d = 0; d < SD; ++d) {
        float xv = xn[d];
        float4 w = *(const float4*)(Wk + (size_t)d * PD + j);
        acc.x += xv * w.x; acc.y += xv * w.y; acc.z += xv * w.z; acc.w += xv * w.w;
    }
    ushort4 o;
    o.x = f2bf(acc.x); o.y = f2bf(acc.y); o.z = f2bf(acc.z); o.w = f2bf(acc.w);
    *(ushort4*)(kmat + (size_t)row * PD + j) = o;
}

// ---------------- kernel 4: Q = tn @ Wq + bq  (bf16 WMMA GEMM, pipelined) -----------
// block = 128 threads = 4 waves; each wave owns a 16(M) x 64(N) tile
__global__ void q_proj_kernel(const unsigned short* __restrict__ tn,
                              const unsigned short* __restrict__ wqT, // [512][1024]
                              const float* __restrict__ bq,
                              unsigned short* __restrict__ q) {
    const int tid   = threadIdx.x;
    const int lane  = tid & 31;
    const int wave  = tid >> 5;
    const int wglob = blockIdx.x * 4 + wave;    // 0 .. 8191
    const int mtile = wglob >> 3;               // 0 .. 1023  (16 rows each)
    const int n0    = (wglob & 7) * 64;         // N block base

    const int m     = lane & 15;
    const int khi   = (lane >> 4) & 1;
    const int nlane = lane & 15;

    v8f acc[4] = {};
    // A fragment source: lane halves hold K {0..7,16..23} / {8..15,24..31}
    const unsigned short* Ap = tn + (size_t)(mtile * 16 + m) * TD + khi * 8;
    // B fragment sources (N-major wqT): lane -> column n, K {0..7,16..23} / {8..15,24..31}
    const unsigned short* Bt[4];
#pragma unroll
    for (int t = 0; t < 4; ++t)
        Bt[t] = wqT + (size_t)(n0 + t * 16 + nlane) * TD + khi * 8;

    // software pipeline: prefetch next K-step's fragments before issuing WMMAs
    u16x8 a0 = *(const u16x8*)(Ap);
    u16x8 a1 = *(const u16x8*)(Ap + 16);
    u16x8 b0[4], b1[4];
#pragma unroll
    for (int t = 0; t < 4; ++t) {
        b0[t] = *(const u16x8*)(Bt[t]);
        b1[t] = *(const u16x8*)(Bt[t] + 16);
    }

    for (int ks = 0; ks < TD / 32; ++ks) {
        const int kn = ((ks + 1) & 31) * 32;    // wraps on last iter (harmless)
        u16x8 na0 = *(const u16x8*)(Ap + kn);
        u16x8 na1 = *(const u16x8*)(Ap + kn + 16);
        u16x8 nb0[4], nb1[4];
#pragma unroll
        for (int t = 0; t < 4; ++t) {
            nb0[t] = *(const u16x8*)(Bt[t] + kn);
            nb1[t] = *(const u16x8*)(Bt[t] + kn + 16);
        }
        v16bf af = pack_frag(a0, a1);
#pragma unroll
        for (int t = 0; t < 4; ++t)
            acc[t] = wmma_bf16(af, pack_frag(b0[t], b1[t]), acc[t]);
        a0 = na0; a1 = na1;
#pragma unroll
        for (int t = 0; t < 4; ++t) { b0[t] = nb0[t]; b1[t] = nb1[t]; }
    }

#pragma unroll
    for (int t = 0; t < 4; ++t) {
        const int n = n0 + t * 16 + nlane;
        const float bias = bq[n];
#pragma unroll
        for (int i = 0; i < 8; ++i) {
            const int row = mtile * 16 + i + 8 * khi;   // C layout: M = i + 8*(lane>=16)
            q[(size_t)row * PD + n] = f2bf(acc[t][i] + bias);
        }
    }
}

// ---------------- kernel 5: scores + softmax + head-mean + entropy partials ----------
// one block per (batch, 16-token tile); 4 waves, wave h = head h.
// K matrix for the batch (128x512 bf16 = 128 KB) is staged into LDS with CDNA5
// async-to-LDS ops; the same LDS is later reused (aliased) for the probs buffer.
__global__ void scores_softmax_kernel(const unsigned short* __restrict__ q,
                                      const unsigned short* __restrict__ kmat,
                                      const float* __restrict__ temperature,
                                      float* __restrict__ out_rw,
                                      float* __restrict__ ent_partials) {
    __shared__ __align__(16) char smem[N_ * PD * 2];   // 128 KB: kbuf, then probs
    __shared__ float entred[128];

    unsigned short* kbuf = (unsigned short*)smem;             // [128][512] bf16
    float (*probs)[16][N_] = (float (*)[16][N_])smem;         // [4][16][128] f32

    const int tid   = threadIdx.x;
    const int lane  = tid & 31;
    const int h     = tid >> 5;              // head
    const int bi    = blockIdx.x >> 8;       // batch  (256 tiles per batch)
    const int stile = blockIdx.x & 255;

    const int m     = lane & 15;
    const int khi   = (lane >> 4) & 1;
    const int nlane = lane & 15;

    // ---- stage K[bi] into LDS via async global->LDS (ASYNCcnt) ----
    const unsigned short* kg = kmat + (size_t)bi * N_ * PD;
    for (int i = tid; i < (N_ * PD) / 8; i += 128)            // 16B chunks
        async_ld16(lds_off_of(kbuf + i * 8), kg + i * 8);
    wait_async0();
    __syncthreads();

    const float tscale = 1.0f / fmaxf(temperature[0], 0.1f);

    v8f acc[8] = {};
    const unsigned short* Ar =
        q + (size_t)(bi * S_ + stile * 16 + m) * PD + h * HD;

#pragma unroll
    for (int ks = 0; ks < HD / 32; ++ks) {   // 4 K-steps
        const int kb = ks * 32;
        u16x8 a0 = *(const u16x8*)(Ar + kb + khi * 8);
        u16x8 a1 = *(const u16x8*)(Ar + kb + 16 + khi * 8);
        v16bf af = pack_frag(a0, a1);
#pragma unroll
        for (int t = 0; t < 8; ++t) {        // 8 state tiles of 16
            const unsigned short* Br =
                kbuf + (size_t)(t * 16 + nlane) * PD + h * HD + kb + khi * 8;
            u16x8 b0 = *(const u16x8*)(Br);
            u16x8 b1 = *(const u16x8*)(Br + 16);
            acc[t] = wmma_bf16(af, pack_frag(b0, b1), acc[t]);
        }
    }

    // per-row softmax over 128 states (register + shfl only):
    // row r lives across 16 lanes x 8 tiles
#pragma unroll
    for (int r = 0; r < 8; ++r) {
        float mx = -3.4e38f;
#pragma unroll
        for (int t = 0; t < 8; ++t) {
            acc[t][r] *= tscale;
            mx = fmaxf(mx, acc[t][r]);
        }
#pragma unroll
        for (int d = 1; d < 16; d <<= 1) mx = fmaxf(mx, __shfl_xor(mx, d, 16));
        float sum = 0.0f;
#pragma unroll
        for (int t = 0; t < 8; ++t) {
            float e = __expf(acc[t][r] - mx);
            acc[t][r] = e;
            sum += e;
        }
#pragma unroll
        for (int d = 1; d < 16; d <<= 1) sum += __shfl_xor(sum, d, 16);
        const float inv = 1.0f / sum;
#pragma unroll
        for (int t = 0; t < 8; ++t) acc[t][r] *= inv;
    }

    // kbuf is dead now; all waves must pass this point before probs overwrites it
    __syncthreads();

#pragma unroll
    for (int r = 0; r < 8; ++r) {
        const int mrow = r + 8 * khi;
#pragma unroll
        for (int t = 0; t < 8; ++t)
            probs[h][mrow][t * 16 + nlane] = acc[t][r];
    }
    __syncthreads();

    // head mean + routing output + entropy partial
    float ent = 0.0f;
#pragma unroll
    for (int j = 0; j < 16; ++j) {
        const int e  = tid * 16 + j;         // 0 .. 2047
        const int mr = e >> 7;               // token within tile
        const int n  = e & 127;              // state index
        const float w = 0.25f * (probs[0][mr][n] + probs[1][mr][n] +
                                 probs[2][mr][n] + probs[3][mr][n]);
        out_rw[(size_t)(bi * S_ + stile * 16 + mr) * N_ + n] = w;
        ent += w * __logf(w + 1e-8f);
    }
    entred[tid] = ent;
    __syncthreads();
    for (int off = 64; off > 0; off >>= 1) {
        if (tid < off) entred[tid] += entred[tid + off];
        __syncthreads();
    }
    if (tid == 0) ent_partials[blockIdx.x] = entred[0];
}

// ---------------- kernel 6: entropy finalize ----------------
__global__ void entropy_finalize_kernel(const float* __restrict__ ent_partials,
                                        float* __restrict__ out_ent) {
    __shared__ float red[256];
    const int tid = threadIdx.x;
    float s = ent_partials[tid] + ent_partials[tid + 256] +
              ent_partials[tid + 512] + ent_partials[tid + 768];
    red[tid] = s;
    __syncthreads();
    for (int off = 128; off > 0; off >>= 1) {
        if (tid < off) red[tid] += red[tid + off];
        __syncthreads();
    }
    // entropy_loss = -mean(entropy)*EW = EW * mean_token( sum_n w*log(w+1e-8) )
    if (tid == 0) out_ent[0] = red[0] * (ENT_W / (float)(B_ * S_));
}

// ---------------- launcher ----------------
extern "C" void kernel_launch(void* const* d_in, const int* in_sizes, int n_in,
                              void* d_out, int out_size, void* d_ws, size_t ws_size,
                              hipStream_t stream) {
    (void)in_sizes; (void)n_in; (void)out_size; (void)ws_size;

    const float* tokens = (const float*)d_in[0];
    const float* states = (const float*)d_in[1];
    const float* ln_t_g = (const float*)d_in[2];
    const float* ln_t_b = (const float*)d_in[3];
    const float* ln_s_g = (const float*)d_in[4];
    const float* ln_s_b = (const float*)d_in[5];
    const float* Wq     = (const float*)d_in[6];
    const float* bq     = (const float*)d_in[7];
    const float* Wk     = (const float*)d_in[8];
    const float* bk     = (const float*)d_in[9];
    const float* temp   = (const float*)d_in[10];

    float* out        = (float*)d_out;
    float* out_tokens = out;
    float* out_rw     = out + TOK_CNT;
    float* out_ent    = out + TOK_CNT + RW_CNT;

    // workspace layout (bytes)
    char* ws = (char*)d_ws;
    unsigned short* tn   = (unsigned short*)(ws);                        // 32 MB
    unsigned short* qbuf = (unsigned short*)(ws + 33554432);             // 16 MB
    unsigned short* kmat = (unsigned short*)(ws + 33554432 + 16777216);  // 512 KB
    unsigned short* wqT  = (unsigned short*)(ws + 33554432 + 16777216 + 524288); // 1 MB
    float* ent_partials  = (float*)(ws + 33554432 + 16777216 + 524288 + 1048576);

    // 1. Wq -> bf16 transposed [N][K]
    wq_transpose_bf16_kernel<<<(PD * TD) / 256, 256, 0, stream>>>(Wq, wqT);
    // 2. LayerNorm tokens (+ passthrough copy)
    ln_tokens_kernel<<<ROWS_T, 256, 0, stream>>>(tokens, ln_t_g, ln_t_b,
                                                 out_tokens, tn);
    // 3. LayerNorm states + K projection
    k_prep_kernel<<<ROWS_K, 128, 0, stream>>>(states, ln_s_g, ln_s_b, Wk, bk, kmat);
    // 4. Q projection (WMMA bf16): 1024 M-tiles x 8 N-blocks = 8192 waves / 4 per block
    q_proj_kernel<<<2048, 128, 0, stream>>>(tn, wqT, bq, qbuf);
    // 5. scores + softmax + head mean + entropy partials: 4 batches x 256 tiles
    scores_softmax_kernel<<<B_ * (S_ / 16), 128, 0, stream>>>(qbuf, kmat, temp,
                                                              out_rw, ent_partials);
    // 6. entropy reduce
    entropy_finalize_kernel<<<1, 256, 0, stream>>>(ent_partials, out_ent);
}
